// AttentionBlock3D_51599737094923
// MI455X (gfx1250) — matrix operand order repaired and checked by reading, not compiled, not run on hardware
//
#include <hip/hip_runtime.h>
#include <hip/hip_bf16.h>

typedef __bf16 bf16;
typedef __attribute__((ext_vector_type(16))) __bf16 v16bf;
typedef __attribute__((ext_vector_type(8)))  float  v8f;

#define NTOK  4096
#define C_DIM 256

union FragU { v16bf v; unsigned int u[8]; };

// A-matrix fragment (16x32 bf16, row-major memory, row stride `ld` elements).
// ISA layout: lane L holds row M=L&15; K = (v&3)*2 + bit + (v>>2)*16 + (L>>4)*8.
__device__ __forceinline__ v16bf load_frag_a(const bf16* base, int ld, int lane) {
  const int m = lane & 15, g = lane >> 4;
  const unsigned int* p = (const unsigned int*)(base + (size_t)m * ld);
  FragU f;
#pragma unroll
  for (int v = 0; v < 8; ++v) {
    const int k = ((v & 3) << 1) + ((v >> 2) << 4) + (g << 3);  // even K of pair
    f.u[v] = p[k >> 1];
  }
  return f.v;
}

// B-matrix fragment (32x16 bf16). Memory is "N-major": element (K,N) at base[N*ld + K];
// lane holds K = (v&3)*2 + bit + (v>>2)*16 + (L>>4)*8, matching the A layout.
__device__ __forceinline__ v16bf load_frag_b(const bf16* base, int ld, int lane) {
  const int n = lane & 15, g = lane >> 4;
  const unsigned int* p = (const unsigned int*)(base + (size_t)n * ld + (g << 3));
  FragU f;
#pragma unroll
  for (int v = 0; v < 8; ++v) f.u[v] = p[(v & 3) + ((v >> 2) << 3)];
  return f.v;
}

__device__ __forceinline__ v8f wmma_bf(v16bf a, v16bf b, v8f c) {
  return __builtin_amdgcn_wmma_f32_16x16x32_bf16(false, a, false, b, (short)0, c,
                                                 false, false);
}

__device__ __forceinline__ float hmax16(float v) {
#pragma unroll
  for (int o = 8; o >= 1; o >>= 1) v = fmaxf(v, __shfl_xor(v, o, 32));
  return v;
}
__device__ __forceinline__ float hsum16(float v) {
#pragma unroll
  for (int o = 8; o >= 1; o >>= 1) v += __shfl_xor(v, o, 32);
  return v;
}

// ---------------- GroupNorm ----------------
// One block per (b, group); a group's 32 channels x 4096 tokens are contiguous.
__global__ void __launch_bounds__(256) gn_stats(const float* __restrict__ x,
                                                float* __restrict__ stats) {
  const int ELEMS = 32 * NTOK;
  const float* p = x + (size_t)blockIdx.x * ELEMS;
  float s = 0.f, s2 = 0.f;
  for (int i = threadIdx.x; i < ELEMS; i += 256) {
    float v = p[i];
    s += v;
    s2 = fmaf(v, v, s2);
  }
  __shared__ float sh1[256], sh2[256];
  sh1[threadIdx.x] = s; sh2[threadIdx.x] = s2;
  __syncthreads();
  for (int o = 128; o > 0; o >>= 1) {
    if (threadIdx.x < o) {
      sh1[threadIdx.x] += sh1[threadIdx.x + o];
      sh2[threadIdx.x] += sh2[threadIdx.x + o];
    }
    __syncthreads();
  }
  if (threadIdx.x == 0) {
    float mean = sh1[0] / (float)ELEMS;
    float var  = sh2[0] / (float)ELEMS - mean * mean;
    stats[2 * blockIdx.x]     = mean;
    stats[2 * blockIdx.x + 1] = rsqrtf(var + 1e-5f);
  }
}

// Normalize + cast to bf16, transposing to token-major [B][N][C].
__global__ void __launch_bounds__(256) gn_apply(const float* __restrict__ x,
                                                const float* __restrict__ stats,
                                                const float* __restrict__ w,
                                                const float* __restrict__ bia,
                                                bf16* __restrict__ xn, int total) {
  int idx = blockIdx.x * blockDim.x + threadIdx.x;
  if (idx >= total) return;
  int n = idx & (NTOK - 1);
  int c = (idx >> 12) & (C_DIM - 1);
  int b = idx >> 20;
  int bg = b * 8 + (c >> 5);
  float v = (x[idx] - stats[2 * bg]) * stats[2 * bg + 1] * w[c] + bia[c];
  xn[((size_t)b * NTOK + n) * C_DIM + c] = (bf16)v;
}

__global__ void __launch_bounds__(256) cvt_bf16(const float* __restrict__ in,
                                                bf16* __restrict__ out, int n) {
  int i = blockIdx.x * blockDim.x + threadIdx.x;
  if (i < n) out[i] = (bf16)in[i];
}

// ---------------- QKV projection GEMM ----------------
// C[o,n] = sum_c W[o,c] * Xn[n,c]; o<512 -> qkbuf token-major, o>=512 -> V d-major.
__device__ __forceinline__ void store_qkv(bf16* qkbuf, bf16* vbuf, int b, int o,
                                          int n, float val) {
  if (o < 512) qkbuf[((size_t)b * NTOK + n) * 512 + o] = (bf16)val;
  else         vbuf[((size_t)b * C_DIM + (o - 512)) * NTOK + n] = (bf16)val;
}

__global__ void __launch_bounds__(256) gemm_qkv(const bf16* __restrict__ W,
                                                const bf16* __restrict__ Xn,
                                                bf16* __restrict__ qkbuf,
                                                bf16* __restrict__ vbuf) {
  const int wave = threadIdx.x >> 5, lane = threadIdx.x & 31;
  const int m0 = blockIdx.x * 64 + (wave >> 2) * 32;
  const int n0 = blockIdx.y * 128 + (wave & 3) * 32;
  const int b  = blockIdx.z;
  const bf16* Xb = Xn + (size_t)b * NTOK * C_DIM;
  v8f a00 = {}, a01 = {}, a10 = {}, a11 = {};
  for (int k0 = 0; k0 < C_DIM; k0 += 32) {
    v16bf fa0 = load_frag_a(W + (size_t)m0 * C_DIM + k0, C_DIM, lane);
    v16bf fa1 = load_frag_a(W + (size_t)(m0 + 16) * C_DIM + k0, C_DIM, lane);
    v16bf fb0 = load_frag_b(Xb + (size_t)n0 * C_DIM + k0, C_DIM, lane);
    v16bf fb1 = load_frag_b(Xb + (size_t)(n0 + 16) * C_DIM + k0, C_DIM, lane);
    a00 = wmma_bf(fa0, fb0, a00);
    a01 = wmma_bf(fa0, fb1, a01);
    a10 = wmma_bf(fa1, fb0, a10);
    a11 = wmma_bf(fa1, fb1, a11);
  }
  const int g = lane >> 4, nn = lane & 15;
#pragma unroll
  for (int r = 0; r < 8; ++r) {
    const int oA = m0 + r + (g << 3), oB = oA + 16;
    const int nA = n0 + nn, nB = nA + 16;
    store_qkv(qkbuf, vbuf, b, oA, nA, a00[r]);
    store_qkv(qkbuf, vbuf, b, oA, nB, a01[r]);
    store_qkv(qkbuf, vbuf, b, oB, nA, a10[r]);
    store_qkv(qkbuf, vbuf, b, oB, nB, a11[r]);
  }
}

// ---------------- Flash attention ----------------
// Block = 8 waves, each wave owns 16 query rows; keys streamed in blocks of 32.
__global__ void __launch_bounds__(256) attn_kernel(const bf16* __restrict__ qk,
                                                   const bf16* __restrict__ vbuf,
                                                   bf16* __restrict__ obuf) {
  __shared__ bf16 pbuf[8][16][32];  // per-wave P-tile for S->A transpose
  const int wave = threadIdx.x >> 5, lane = threadIdx.x & 31;
  const int b = blockIdx.z, h = blockIdx.y;
  const int q0 = blockIdx.x * 128 + wave * 16;

  const bf16* Qb = qk + ((size_t)b * NTOK) * 512 + h * 32;  // q[t][d] at t*512+d
  const bf16* Kb = Qb + 256;                                // k[t][d]
  const bf16* Vb = vbuf + ((size_t)b * C_DIM + h * 32) * NTOK;  // v[d][t]

  const v16bf qa = load_frag_a(Qb + (size_t)q0 * 512, 512, lane);

  v8f oacc0 = {}, oacc1 = {};
  float row_m[8], row_l[8];
#pragma unroll
  for (int r = 0; r < 8; ++r) { row_m[r] = -1e30f; row_l[r] = 0.f; }

  const float scale = 0.1767766952966369f;  // 32^-0.5

  for (int t0 = 0; t0 < NTOK; t0 += 32) {
    v16bf kb0 = load_frag_b(Kb + (size_t)t0 * 512, 512, lane);
    v16bf kb1 = load_frag_b(Kb + (size_t)(t0 + 16) * 512, 512, lane);
    if (t0 + 32 < NTOK) {
      __builtin_prefetch(Kb + (size_t)(t0 + 32) * 512, 0, 1);
      __builtin_prefetch(Vb + t0 + 32, 0, 1);
    }
    v8f s0 = {}, s1 = {};
    s0 = wmma_bf(qa, kb0, s0);  // scores for keys t0..t0+15
    s1 = wmma_bf(qa, kb1, s1);  // keys t0+16..t0+31

    float alpha[8];
#pragma unroll
    for (int r = 0; r < 8; ++r) {
      float sa = s0[r] * scale, sc = s1[r] * scale;
      float m = fmaxf(row_m[r], hmax16(fmaxf(sa, sc)));
      float p0 = __expf(sa - m), p1 = __expf(sc - m);
      float sum = hsum16(p0 + p1);
      alpha[r] = __expf(row_m[r] - m);
      row_l[r] = row_l[r] * alpha[r] + sum;
      row_m[r] = m;
      const int q = r + ((lane >> 4) << 3);
      const int t = lane & 15;
      pbuf[wave][q][t]      = (bf16)p0;
      pbuf[wave][q][t + 16] = (bf16)p1;
    }
#pragma unroll
    for (int r = 0; r < 8; ++r) { oacc0[r] *= alpha[r]; oacc1[r] *= alpha[r]; }
    __syncthreads();
    const v16bf pa  = load_frag_a(&pbuf[wave][0][0], 32, lane);
    const v16bf vb0 = load_frag_b(Vb + t0, NTOK, lane);                      // d 0..15
    const v16bf vb1 = load_frag_b(Vb + (size_t)16 * NTOK + t0, NTOK, lane);  // d 16..31
    oacc0 = wmma_bf(pa, vb0, oacc0);
    oacc1 = wmma_bf(pa, vb1, oacc1);
    __syncthreads();
  }

  const int g = lane >> 4, dd = lane & 15;
#pragma unroll
  for (int r = 0; r < 8; ++r) {
    const int q = q0 + r + (g << 3);
    const float inv = 1.f / row_l[r];
    bf16* o = obuf + ((size_t)b * NTOK + q) * C_DIM + h * 32 + dd;
    o[0]  = (bf16)(oacc0[r] * inv);
    o[16] = (bf16)(oacc1[r] * inv);
  }
}

// ---------------- Output projection + bias + residual ----------------
__global__ void __launch_bounds__(256) gemm_out(const bf16* __restrict__ W,
                                                const bf16* __restrict__ A,
                                                const float* __restrict__ bias,
                                                const float* __restrict__ x,
                                                float* __restrict__ out) {
  const int wave = threadIdx.x >> 5, lane = threadIdx.x & 31;
  const int m0 = blockIdx.x * 64 + (wave >> 2) * 32;
  const int n0 = blockIdx.y * 128 + (wave & 3) * 32;
  const int b  = blockIdx.z;
  const bf16* Ab = A + (size_t)b * NTOK * C_DIM;
  v8f a00 = {}, a01 = {}, a10 = {}, a11 = {};
  for (int k0 = 0; k0 < C_DIM; k0 += 32) {
    v16bf fa0 = load_frag_a(W + (size_t)m0 * C_DIM + k0, C_DIM, lane);
    v16bf fa1 = load_frag_a(W + (size_t)(m0 + 16) * C_DIM + k0, C_DIM, lane);
    v16bf fb0 = load_frag_b(Ab + (size_t)n0 * C_DIM + k0, C_DIM, lane);
    v16bf fb1 = load_frag_b(Ab + (size_t)(n0 + 16) * C_DIM + k0, C_DIM, lane);
    a00 = wmma_bf(fa0, fb0, a00);
    a01 = wmma_bf(fa0, fb1, a01);
    a10 = wmma_bf(fa1, fb0, a10);
    a11 = wmma_bf(fa1, fb1, a11);
  }
  const int g = lane >> 4, nn = lane & 15;
#pragma unroll
  for (int r = 0; r < 8; ++r) {
    const int oA = m0 + r + (g << 3), oB = oA + 16;
    const int nA = n0 + nn, nB = nA + 16;
    size_t i00 = ((size_t)b * C_DIM + oA) * NTOK + nA;
    size_t i01 = ((size_t)b * C_DIM + oA) * NTOK + nB;
    size_t i10 = ((size_t)b * C_DIM + oB) * NTOK + nA;
    size_t i11 = ((size_t)b * C_DIM + oB) * NTOK + nB;
    out[i00] = a00[r] + bias[oA] + x[i00];
    out[i01] = a01[r] + bias[oA] + x[i01];
    out[i10] = a10[r] + bias[oB] + x[i10];
    out[i11] = a11[r] + bias[oB] + x[i11];
  }
}

extern "C" void kernel_launch(void* const* d_in, const int* in_sizes, int n_in,
                              void* d_out, int out_size, void* d_ws, size_t ws_size,
                              hipStream_t stream) {
  (void)n_in; (void)out_size; (void)ws_size;
  const float* x      = (const float*)d_in[0];
  const float* norm_w = (const float*)d_in[1];
  const float* norm_b = (const float*)d_in[2];
  const float* w_qkv  = (const float*)d_in[3];
  const float* w_out  = (const float*)d_in[4];
  const float* b_out  = (const float*)d_in[5];
  float* out = (float*)d_out;

  const int B = in_sizes[0] / (C_DIM * NTOK);  // 2

  char* ws = (char*)d_ws;
  float* stats  = (float*)ws;  ws += 256;                          // [B*8][mean,rstd]
  bf16* wqkv_bf = (bf16*)ws;   ws += (size_t)768 * C_DIM * 2;      // [768][256]
  bf16* wout_bf = (bf16*)ws;   ws += (size_t)C_DIM * C_DIM * 2;    // [256][256]
  bf16* xn      = (bf16*)ws;   ws += (size_t)B * NTOK * C_DIM * 2; // [B][N][C]
  bf16* qkbuf   = (bf16*)ws;   ws += (size_t)B * NTOK * 512 * 2;   // [B][N][512]
  bf16* vbuf    = (bf16*)ws;   ws += (size_t)B * C_DIM * NTOK * 2; // [B][256][N]
  bf16* obuf    = (bf16*)ws;                                       // [B][N][C]

  gn_stats<<<B * 8, 256, 0, stream>>>(x, stats);
  const int total = B * C_DIM * NTOK;
  gn_apply<<<(total + 255) / 256, 256, 0, stream>>>(x, stats, norm_w, norm_b, xn, total);
  cvt_bf16<<<(768 * C_DIM) / 256, 256, 0, stream>>>(w_qkv, wqkv_bf, 768 * C_DIM);
  cvt_bf16<<<(C_DIM * C_DIM) / 256, 256, 0, stream>>>(w_out, wout_bf, C_DIM * C_DIM);
  gemm_qkv<<<dim3(768 / 64, NTOK / 128, B), 256, 0, stream>>>(wqkv_bf, xn, qkbuf, vbuf);
  attn_kernel<<<dim3(NTOK / 128, 8, B), 256, 0, stream>>>(qkbuf, vbuf, obuf);
  gemm_out<<<dim3(C_DIM / 64, NTOK / 128, B), 256, 0, stream>>>(wout_bf, obuf, b_out, x, out);
}
